// EGNNPocketLayer_34084860461598
// MI455X (gfx1250) — compile-verified, matrix operand order checked by reading, not run on hardware
//
#include <hip/hip_runtime.h>
#include <hip/hip_bf16.h>

typedef __attribute__((ext_vector_type(16))) _Float16 v16h;
typedef __attribute__((ext_vector_type(8)))  _Float16 v8h;
typedef __attribute__((ext_vector_type(4)))  _Float16 v4h;
typedef __attribute__((ext_vector_type(8)))  float    v8f;

#define HDIM   128
#define NTILES 8      // 8 n-tiles of 16 cols = 128 outputs
#define WAVES  4
#define TILE_M 16

union V16U { v16h v; v8h h8[2]; };

__device__ __forceinline__ float silu_f(float x) { return x / (1.0f + __expf(-x)); }

// Wave-internal LDS producer->consumer fence (CDNA5 split counters).
__device__ __forceinline__ void wave_lds_fence() {
  __builtin_amdgcn_wave_barrier();
  asm volatile("s_wait_dscnt 0x0" ::: "memory");
  __builtin_amdgcn_wave_barrier();
}

// A-fragment (16x32 f16) per documented CDNA5 layout:
// lane L: row m = L&15, sel = L>>4; halfs K = sel*8 + {0..7} and sel*8 + {16..23}.
__device__ __forceinline__ v16h load_afrag(const _Float16* base, int stride, int kk, int lane) {
  const int m = lane & 15, sel = lane >> 4;
  const _Float16* p = base + m * stride + kk * 32 + sel * 8;
  V16U u;
  u.h8[0] = *(const v8h*)(p);
  u.h8[1] = *(const v8h*)(p + 16);
  return u.v;
}

// B-fragment pre-packed in global memory: 32 bytes contiguous per lane per fragment.
__device__ __forceinline__ v16h load_bfrag(const _Float16* wp, int frag, int lane) {
  return *(const v16h*)(wp + (((size_t)frag * 32 + lane) << 4));
}

template <int KC>
__device__ __forceinline__ void gemm_tile(const _Float16* A, int stride,
                                          const _Float16* Wp, int lane, v8f acc[NTILES]) {
#pragma unroll
  for (int kk = 0; kk < KC; ++kk) {
    v16h a = load_afrag(A, stride, kk, lane);
#pragma unroll
    for (int nt = 0; nt < NTILES; ++nt) {
      v16h b = load_bfrag(Wp, kk * NTILES + nt, lane);
      acc[nt] = __builtin_amdgcn_wmma_f32_16x16x32_f16(
          false, a, false, b, (short)0, acc[nt], false, false);
    }
  }
}

// Pack weight W[128, rowStride] (row-major, out x in) into per-lane B-fragment order (f16).
__global__ void egnn_pack_b(const float* __restrict__ W, _Float16* __restrict__ dst,
                            int rowStride, int total) {
  int idx = blockIdx.x * blockDim.x + threadIdx.x;
  if (idx >= total) return;
  int hh  = idx & 15;
  int L   = (idx >> 4) & 31;
  int fr  = idx >> 9;
  int nt  = fr & 7, kk = fr >> 3;
  int nu  = L & 15, sel = L >> 4;
  int kap = sel * 8 + (hh & 7) + ((hh & 8) ? 16 : 0);
  int n = nt * 16 + nu;
  int k = kk * 32 + kap;
  dst[idx] = (_Float16)W[(size_t)n * rowStride + k];
}

__global__ __launch_bounds__(128) void egnn_edge_kernel(
    const float* __restrict__ h, const float* __restrict__ pos,
    const int* __restrict__ ei, long long E, long long etiles,
    const float* __restrict__ We1, const float* __restrict__ be1,
    const float* __restrict__ be2, const float* __restrict__ bc1,
    const float* __restrict__ Wc2, const float* __restrict__ bc2,
    const _Float16* __restrict__ We1p, const _Float16* __restrict__ We2p,
    const _Float16* __restrict__ Wc1p,
    float* __restrict__ agg, float* __restrict__ posOut) {
  __shared__ _Float16 ldsA[WAVES][TILE_M * 256];
  __shared__ _Float16 ldsM[WAVES][TILE_M * 128];
  __shared__ float ldsRad[WAVES][TILE_M][4];
  __shared__ int   ldsRow[WAVES][TILE_M];
  __shared__ int   ldsCol[WAVES][TILE_M];
  __shared__ float ldsS[WAVES][TILE_M];
  __shared__ float sWcol[HDIM], sB1[HDIM], sB2[HDIM], sBC1[HDIM], sWc2[HDIM];

  const int wid  = threadIdx.x >> 5;
  const int lane = threadIdx.x & 31;
  const int nu = lane & 15, sel = lane >> 4;
  const long long tile = (long long)blockIdx.x * WAVES + wid;
  if (tile >= etiles) return;  // exited waves drop out of barrier membership
  const long long ebase = tile * TILE_M;

  for (int i = threadIdx.x; i < HDIM; i += blockDim.x) {
    sWcol[i] = We1[(size_t)i * 257 + 256];
    sB1[i] = be1[i]; sB2[i] = be2[i]; sBC1[i] = bc1[i]; sWc2[i] = Wc2[i];
  }
  if (lane < TILE_M) {
    long long e = ebase + lane;
    int r = ei[e], c = ei[E + e];
    ldsRow[wid][lane] = r; ldsCol[wid][lane] = c;
    float rx = pos[(size_t)r * 3 + 0] - pos[(size_t)c * 3 + 0];
    float ry = pos[(size_t)r * 3 + 1] - pos[(size_t)c * 3 + 1];
    float rz = pos[(size_t)r * 3 + 2] - pos[(size_t)c * 3 + 2];
    float rn = fmaxf(sqrtf(rx * rx + ry * ry + rz * rz), 1e-8f);
    ldsRad[wid][lane][0] = rx; ldsRad[wid][lane][1] = ry;
    ldsRad[wid][lane][2] = rz; ldsRad[wid][lane][3] = rn;
  }
  __syncthreads();

  // Gather h[row] | h[col] -> LDS f16 (A-matrix staging, rows of 256 halfs)
#pragma unroll 4
  for (int m = 0; m < TILE_M; ++m) {
    int r = ldsRow[wid][m], c = ldsCol[wid][m];
    float4 hr = *(const float4*)(h + (size_t)r * HDIM + lane * 4);
    float4 hc = *(const float4*)(h + (size_t)c * HDIM + lane * 4);
    v4h pr = {(_Float16)hr.x, (_Float16)hr.y, (_Float16)hr.z, (_Float16)hr.w};
    v4h pc = {(_Float16)hc.x, (_Float16)hc.y, (_Float16)hc.z, (_Float16)hc.w};
    *(v4h*)(&ldsA[wid][m * 256 + lane * 4])       = pr;
    *(v4h*)(&ldsA[wid][m * 256 + 128 + lane * 4]) = pc;
  }
  __syncthreads();

  // ---- GEMM1: silu(edge_input @ We1^T + be1), K=256 via WMMA + rank-1 radial term
  v8f acc[NTILES];
#pragma unroll
  for (int nt = 0; nt < NTILES; ++nt) {
    float bv = sB1[nt * 16 + nu];
#pragma unroll
    for (int i = 0; i < 8; ++i) acc[nt][i] = bv;
  }
  gemm_tile<8>(ldsA[wid], 256, We1p, lane, acc);
#pragma unroll
  for (int nt = 0; nt < NTILES; ++nt) {
    float w = sWcol[nt * 16 + nu];
#pragma unroll
    for (int r = 0; r < 8; ++r) {
      int m = r + (sel << 3);
      float v = silu_f(acc[nt][r] + w * ldsRad[wid][m][3]);
      ldsM[wid][m * 128 + nt * 16 + nu] = (_Float16)v;
    }
  }
  __syncthreads();

  // ---- GEMM2: m_ij = a1 @ We2^T + be2, K=128
#pragma unroll
  for (int nt = 0; nt < NTILES; ++nt) {
    float bv = sB2[nt * 16 + nu];
#pragma unroll
    for (int i = 0; i < 8; ++i) acc[nt][i] = bv;
  }
  gemm_tile<4>(ldsM[wid], 128, We2p, lane, acc);

  // segment-sum into agg (f32 global atomics) + stage m_ij as f16 for coord MLP
#pragma unroll
  for (int nt = 0; nt < NTILES; ++nt) {
    int n = nt * 16 + nu;
#pragma unroll
    for (int r = 0; r < 8; ++r) {
      int m = r + (sel << 3);
      float v = acc[nt][r];
      unsafeAtomicAdd(&agg[(size_t)ldsRow[wid][m] * HDIM + n], v);
      ldsM[wid][m * 128 + n] = (_Float16)v;
    }
  }
  __syncthreads();

  // ---- GEMM3: t = silu(m_ij @ Wc1^T + bc1), K=128
#pragma unroll
  for (int nt = 0; nt < NTILES; ++nt) {
    float bv = sBC1[nt * 16 + nu];
#pragma unroll
    for (int i = 0; i < 8; ++i) acc[nt][i] = bv;
  }
  gemm_tile<4>(ldsM[wid], 128, Wc1p, lane, acc);

  // dot with Wc2 (Linear(H,1)) via LDS f32 atomics
  if (lane < TILE_M) ldsS[wid][lane] = 0.0f;
  __syncthreads();
  float part[8];
#pragma unroll
  for (int r = 0; r < 8; ++r) part[r] = 0.0f;
#pragma unroll
  for (int nt = 0; nt < NTILES; ++nt) {
    float w = sWc2[nt * 16 + nu];
#pragma unroll
    for (int r = 0; r < 8; ++r) part[r] += silu_f(acc[nt][r]) * w;
  }
#pragma unroll
  for (int r = 0; r < 8; ++r) atomicAdd(&ldsS[wid][r + (sel << 3)], part[r]);
  __syncthreads();

  if (lane < TILE_M) {
    float g = tanhf(ldsS[wid][lane] + bc2[0]) * 0.1f;
    float scale = g / ldsRad[wid][lane][3];
    int rr = ldsRow[wid][lane];
    unsafeAtomicAdd(&posOut[(size_t)rr * 3 + 0], ldsRad[wid][lane][0] * scale);
    unsafeAtomicAdd(&posOut[(size_t)rr * 3 + 1], ldsRad[wid][lane][1] * scale);
    unsafeAtomicAdd(&posOut[(size_t)rr * 3 + 2], ldsRad[wid][lane][2] * scale);
  }
}

__global__ __launch_bounds__(128) void egnn_node_kernel(
    const float* __restrict__ h, const float* __restrict__ agg,
    const float* __restrict__ bn1, const float* __restrict__ bn2,
    const _Float16* __restrict__ Wn1p, const _Float16* __restrict__ Wn2p,
    float* __restrict__ hOut, int ntiles) {
  __shared__ _Float16 ldsA[WAVES][TILE_M * 256];
  __shared__ _Float16 ldsM[WAVES][TILE_M * 128];
  const int wid  = threadIdx.x >> 5;
  const int lane = threadIdx.x & 31;
  const int nu = lane & 15, sel = lane >> 4;
  int tile = blockIdx.x * WAVES + wid;
  if (tile >= ntiles) return;
  size_t base = (size_t)tile * TILE_M;

#pragma unroll 4
  for (int m = 0; m < TILE_M; ++m) {
    float4 hv = *(const float4*)(h   + (base + m) * HDIM + lane * 4);
    float4 av = *(const float4*)(agg + (base + m) * HDIM + lane * 4);
    v4h p1 = {(_Float16)hv.x, (_Float16)hv.y, (_Float16)hv.z, (_Float16)hv.w};
    v4h p2 = {(_Float16)av.x, (_Float16)av.y, (_Float16)av.z, (_Float16)av.w};
    *(v4h*)(&ldsA[wid][m * 256 + lane * 4])       = p1;
    *(v4h*)(&ldsA[wid][m * 256 + 128 + lane * 4]) = p2;
  }
  wave_lds_fence();

  v8f acc[NTILES];
#pragma unroll
  for (int nt = 0; nt < NTILES; ++nt) {
    float bv = bn1[nt * 16 + nu];
#pragma unroll
    for (int i = 0; i < 8; ++i) acc[nt][i] = bv;
  }
  gemm_tile<8>(ldsA[wid], 256, Wn1p, lane, acc);
#pragma unroll
  for (int nt = 0; nt < NTILES; ++nt) {
#pragma unroll
    for (int r = 0; r < 8; ++r) {
      int m = r + (sel << 3);
      ldsM[wid][m * 128 + nt * 16 + nu] = (_Float16)silu_f(acc[nt][r]);
    }
  }
  wave_lds_fence();

#pragma unroll
  for (int nt = 0; nt < NTILES; ++nt) {
    float bv = bn2[nt * 16 + nu];
#pragma unroll
    for (int i = 0; i < 8; ++i) acc[nt][i] = bv;
  }
  gemm_tile<4>(ldsM[wid], 128, Wn2p, lane, acc);

#pragma unroll
  for (int nt = 0; nt < NTILES; ++nt) {
    int n = nt * 16 + nu;
#pragma unroll
    for (int r = 0; r < 8; ++r) {
      int m = r + (sel << 3);
      size_t o = (base + m) * HDIM + n;
      hOut[o] = h[o] + acc[nt][r];
    }
  }
}

extern "C" void kernel_launch(void* const* d_in, const int* in_sizes, int n_in,
                              void* d_out, int out_size, void* d_ws, size_t ws_size,
                              hipStream_t stream) {
  const float* h   = (const float*)d_in[0];
  const float* pos = (const float*)d_in[1];
  const int*   ei  = (const int*)d_in[2];
  const float* We1 = (const float*)d_in[3];
  const float* be1 = (const float*)d_in[4];
  const float* We2 = (const float*)d_in[5];
  const float* be2 = (const float*)d_in[6];
  const float* Wn1 = (const float*)d_in[7];
  const float* bn1 = (const float*)d_in[8];
  const float* Wn2 = (const float*)d_in[9];
  const float* bn2 = (const float*)d_in[10];
  const float* Wc1 = (const float*)d_in[11];
  const float* bc1 = (const float*)d_in[12];
  const float* Wc2 = (const float*)d_in[13];
  const float* bc2 = (const float*)d_in[14];

  const long long N = in_sizes[0] / HDIM;
  const long long E = in_sizes[2] / 2;

  float* hOut   = (float*)d_out;
  float* posOut = hOut + (size_t)N * HDIM;

  char* ws = (char*)d_ws;
  size_t off = 0;
  float* agg = (float*)(ws + off);
  off += (size_t)N * HDIM * sizeof(float);
  off = (off + 255) & ~(size_t)255;
  _Float16* We1p = (_Float16*)(ws + off); off += 32768 * sizeof(_Float16);
  _Float16* We2p = (_Float16*)(ws + off); off += 16384 * sizeof(_Float16);
  _Float16* Wn1p = (_Float16*)(ws + off); off += 32768 * sizeof(_Float16);
  _Float16* Wn2p = (_Float16*)(ws + off); off += 16384 * sizeof(_Float16);
  _Float16* Wc1p = (_Float16*)(ws + off); off += 16384 * sizeof(_Float16);

  hipMemsetAsync(agg, 0, (size_t)N * HDIM * sizeof(float), stream);
  hipMemcpyAsync(posOut, pos, (size_t)N * 3 * sizeof(float),
                 hipMemcpyDeviceToDevice, stream);

  egnn_pack_b<<<(32768 + 255) / 256, 256, 0, stream>>>(We1, We1p, 257, 32768);
  egnn_pack_b<<<(16384 + 255) / 256, 256, 0, stream>>>(We2, We2p, 128, 16384);
  egnn_pack_b<<<(32768 + 255) / 256, 256, 0, stream>>>(Wn1, Wn1p, 256, 32768);
  egnn_pack_b<<<(16384 + 255) / 256, 256, 0, stream>>>(Wn2, Wn2p, 128, 16384);
  egnn_pack_b<<<(16384 + 255) / 256, 256, 0, stream>>>(Wc1, Wc1p, 128, 16384);

  long long etiles = (E + TILE_M - 1) / TILE_M;
  int eblocks = (int)((etiles + WAVES - 1) / WAVES);
  egnn_edge_kernel<<<eblocks, WAVES * 32, 0, stream>>>(
      h, pos, ei, E, etiles, We1, be1, be2, bc1, Wc2, bc2,
      We1p, We2p, Wc1p, agg, posOut);

  int ntilesN = (int)((N + TILE_M - 1) / TILE_M);
  int nblocks = (ntilesN + WAVES - 1) / WAVES;
  egnn_node_kernel<<<nblocks, WAVES * 32, 0, stream>>>(
      h, agg, bn1, bn2, Wn1p, Wn2p, hOut, ntilesN);
}
